// PointConvDensitySetInterpolation_9242769621825
// MI455X (gfx1250) — matrix operand order repaired and checked by reading, not compiled, not run on hardware
//
#include <hip/hip_runtime.h>
#include <stdint.h>
#include <stddef.h>

// ---------------- problem constants ----------------
#define Bb 8
#define Nn 2048
#define Ss 512
#define Dd 256
#define KNN 16
#define COLS (KNN*Nn)       // 32768 columns per batch for the mlp/wn nets
#define FCOLS (Bb*Nn)       // 16384 columns for the final linear
#define CIN0 259
#define CIN0P 288           // 259 padded to multiple of 32
#define CMID 128
#define COUT3 256
#define LIN_IN 4096
#define LIN_OUT 256
#define EPS 1e-5f

typedef __bf16 bf16t;
typedef __attribute__((ext_vector_type(16))) bf16t v16bf;
typedef __attribute__((ext_vector_type(8)))  float v8f;

// ---------------- helpers ----------------
__device__ __forceinline__ unsigned short f2bf(float f) {
  unsigned int u = __float_as_uint(f);
  u = u + 0x7FFFu + ((u >> 16) & 1u);
  return (unsigned short)(u >> 16);
}
__device__ __forceinline__ float bf2f(unsigned short h) {
  return __uint_as_float(((unsigned int)h) << 16);
}
__device__ __forceinline__ float waveSum(float v) {
#pragma unroll
  for (int o = 16; o > 0; o >>= 1) v += __shfl_xor(v, o, 32);
  return v;
}
// keep the 3 smallest (d,i)
__device__ __forceinline__ void ins3(float d, int i,
                                     float& d0, int& i0, float& d1, int& i1,
                                     float& d2, int& i2) {
  if (d < d2) {
    if (d < d1) {
      d2 = d1; i2 = i1;
      if (d < d0) { d1 = d0; i1 = i0; d0 = d; i0 = i; }
      else        { d1 = d;  i1 = i; }
    } else { d2 = d; i2 = i; }
  }
}

// ---------------- weight prep: f32 -> bf16 (padded) ----------------
__global__ void prep_weights_kernel(const float* __restrict__ w0,
                                    const float* __restrict__ w1,
                                    const float* __restrict__ w2,
                                    const float* __restrict__ wl,
                                    unsigned short* __restrict__ w0p,
                                    unsigned short* __restrict__ w1p,
                                    unsigned short* __restrict__ w2p,
                                    unsigned short* __restrict__ wlp) {
  int idx = blockIdx.x * blockDim.x + threadIdx.x;
  int total = LIN_OUT * LIN_IN;           // 1,048,576 (largest)
  for (; idx < total; idx += gridDim.x * blockDim.x) {
    if (idx < CMID * CIN0P) {
      int r = idx / CIN0P, c = idx % CIN0P;
      w0p[idx] = (c < CIN0) ? f2bf(w0[r * CIN0 + c]) : (unsigned short)0;
    }
    if (idx < CMID * CMID)  w1p[idx] = f2bf(w1[idx]);
    if (idx < COUT3 * CMID) w2p[idx] = f2bf(w2[idx]);
    wlp[idx] = f2bf(wl[idx]);
  }
}

// ---------------- transpose (B,3,N)->(B,N,3) ----------------
__global__ void transpose_kernel(const float* __restrict__ xyz1,
                                 const float* __restrict__ xyz2,
                                 float* __restrict__ xyz1t,
                                 float* __restrict__ xyz2t) {
  int idx = blockIdx.x * blockDim.x + threadIdx.x;
  int t1 = Bb * 3 * Nn, t2 = Bb * 3 * Ss;
  for (; idx < t1 + t2; idx += gridDim.x * blockDim.x) {
    if (idx < t1) {
      int b = idx / (3 * Nn), r = idx % (3 * Nn);
      int d = r / Nn, n = r % Nn;
      xyz1t[(b * Nn + n) * 3 + d] = xyz1[idx];
    } else {
      int j = idx - t1;
      int b = j / (3 * Ss), r = j % (3 * Ss);
      int d = r / Ss, s = r % Ss;
      xyz2t[(b * Ss + s) * 3 + d] = xyz2[j];
    }
  }
}

// ---------------- 3-NN inverse-distance interpolation ----------------
// one wave per (b,n); output interp[(b*N+n)*D + c] bf16
__global__ void interp_kernel(const float* __restrict__ xyz1t,
                              const float* __restrict__ xyz2t,
                              const float* __restrict__ points2,
                              unsigned short* __restrict__ interp) {
  int wid = threadIdx.x >> 5, lane = threadIdx.x & 31;
  int q = blockIdx.x * 8 + wid;
  if (q >= Bb * Nn) return;
  int b = q / Nn;
  float qx = xyz1t[q * 3 + 0], qy = xyz1t[q * 3 + 1], qz = xyz1t[q * 3 + 2];
  float d0 = 3.4e38f, d1 = 3.4e38f, d2 = 3.4e38f;
  int   i0 = 0, i1 = 0, i2 = 0;
  for (int s = lane; s < Ss; s += 32) {
    const float* p = &xyz2t[(b * Ss + s) * 3];
    float dx = qx - p[0], dy = qy - p[1], dz = qz - p[2];
    ins3(dx * dx + dy * dy + dz * dz, s, d0, i0, d1, i1, d2, i2);
  }
#pragma unroll
  for (int off = 16; off > 0; off >>= 1) {
    float od0 = __shfl_xor(d0, off, 32), od1 = __shfl_xor(d1, off, 32), od2 = __shfl_xor(d2, off, 32);
    int   oi0 = __shfl_xor(i0, off, 32), oi1 = __shfl_xor(i1, off, 32), oi2 = __shfl_xor(i2, off, 32);
    ins3(od0, oi0, d0, i0, d1, i1, d2, i2);
    ins3(od1, oi1, d0, i0, d1, i1, d2, i2);
    ins3(od2, oi2, d0, i0, d1, i1, d2, i2);
  }
  float r0 = 1.0f / (d0 + 1e-8f), r1 = 1.0f / (d1 + 1e-8f), r2 = 1.0f / (d2 + 1e-8f);
  float inv = 1.0f / (r0 + r1 + r2);
  float w0 = r0 * inv, w1 = r1 * inv, w2 = r2 * inv;
  for (int c = lane; c < Dd; c += 32) {
    const float* pc = &points2[((size_t)b * Dd + c) * Ss];
    float v = w0 * pc[i0] + w1 * pc[i1] + w2 * pc[i2];
    interp[(size_t)q * Dd + c] = f2bf(v);
  }
}

// ---------------- kernel density: mean_m exp(-d^2/2)/2.5 ----------------
__global__ void density_kernel(const float* __restrict__ xyz1t,
                               float* __restrict__ density) {
  int wid = threadIdx.x >> 5, lane = threadIdx.x & 31;
  int q = blockIdx.x * 8 + wid;
  if (q >= Bb * Nn) return;
  int b = q / Nn;
  float qx = xyz1t[q * 3 + 0], qy = xyz1t[q * 3 + 1], qz = xyz1t[q * 3 + 2];
  float s = 0.0f;
  for (int m = lane; m < Nn; m += 32) {
    const float* p = &xyz1t[((size_t)b * Nn + m) * 3];
    float dx = qx - p[0], dy = qy - p[1], dz = qz - p[2];
    s += __expf(-0.5f * (dx * dx + dy * dy + dz * dz));
  }
  s = waveSum(s);
  if (lane == 0) density[q] = (s * (1.0f / 2.5f)) / (float)Nn;
}

// ---------------- density net (1->8->8->1, inorm over N) ----------------
__global__ void dn_net_kernel(const float* __restrict__ density,
                              const float* __restrict__ w0, const float* __restrict__ b0,
                              const float* __restrict__ w1, const float* __restrict__ b1,
                              const float* __restrict__ w2, const float* __restrict__ b2,
                              float* __restrict__ density_scale) {
  __shared__ unsigned short lA[8][Nn];
  __shared__ unsigned short lB[8][Nn];
  int b = blockIdx.x;
  int tid = threadIdx.x, lane = tid & 31, wid = tid >> 5;

  for (int n = tid; n < Nn; n += 256) {
    float x = density[b * Nn + n];
#pragma unroll
    for (int c = 0; c < 8; ++c) lA[c][n] = f2bf(w0[c] * x + b0[c]);
  }
  __syncthreads();
  { // inorm+relu on lA, channel = wid
    float s1 = 0.f, s2 = 0.f;
    for (int n = lane; n < Nn; n += 32) { float x = bf2f(lA[wid][n]); s1 += x; s2 += x * x; }
    s1 = waveSum(s1); s2 = waveSum(s2);
    float m = s1 / (float)Nn, v = s2 / (float)Nn - m * m, iv = rsqrtf(v + EPS);
    for (int n = lane; n < Nn; n += 32) {
      float x = (bf2f(lA[wid][n]) - m) * iv;
      lA[wid][n] = f2bf(fmaxf(x, 0.0f));
    }
  }
  __syncthreads();
  for (int n = tid; n < Nn; n += 256) {
    float xi[8];
#pragma unroll
    for (int i = 0; i < 8; ++i) xi[i] = bf2f(lA[i][n]);
#pragma unroll
    for (int c = 0; c < 8; ++c) {
      float s = b1[c];
#pragma unroll
      for (int i = 0; i < 8; ++i) s += w1[c * 8 + i] * xi[i];
      lB[c][n] = f2bf(s);
    }
  }
  __syncthreads();
  { // inorm+relu on lB
    float s1 = 0.f, s2 = 0.f;
    for (int n = lane; n < Nn; n += 32) { float x = bf2f(lB[wid][n]); s1 += x; s2 += x * x; }
    s1 = waveSum(s1); s2 = waveSum(s2);
    float m = s1 / (float)Nn, v = s2 / (float)Nn - m * m, iv = rsqrtf(v + EPS);
    for (int n = lane; n < Nn; n += 32) {
      float x = (bf2f(lB[wid][n]) - m) * iv;
      lB[wid][n] = f2bf(fmaxf(x, 0.0f));
    }
  }
  __syncthreads();
  for (int n = tid; n < Nn; n += 256) {
    float s = b2[0];
#pragma unroll
    for (int i = 0; i < 8; ++i) s += w2[i] * bf2f(lB[i][n]);
    lA[0][n] = f2bf(s);
  }
  __syncthreads();
  if (wid == 0) { // final 1-channel inorm+relu
    float s1 = 0.f, s2 = 0.f;
    for (int n = lane; n < Nn; n += 32) { float x = bf2f(lA[0][n]); s1 += x; s2 += x * x; }
    s1 = waveSum(s1); s2 = waveSum(s2);
    float m = s1 / (float)Nn, v = s2 / (float)Nn - m * m, iv = rsqrtf(v + EPS);
    for (int n = lane; n < Nn; n += 32) {
      float x = (bf2f(lA[0][n]) - m) * iv;
      density_scale[b * Nn + n] = fmaxf(x, 0.0f);
    }
  }
}

// ---------------- farthest point sampling (sequential) ----------------
__global__ void fps_kernel(const float* __restrict__ xyz1t, int* __restrict__ fps_idx) {
  __shared__ float px[Nn], py[Nn], pz[Nn], dist[Nn];
  __shared__ float wmx[8];
  __shared__ int wmi[8];
  __shared__ int s_far;
  int b = blockIdx.x;
  int tid = threadIdx.x, lane = tid & 31, wid = tid >> 5;
  for (int n = tid; n < Nn; n += 256) {
    const float* p = &xyz1t[((size_t)b * Nn + n) * 3];
    px[n] = p[0]; py[n] = p[1]; pz[n] = p[2];
    dist[n] = 1e10f;
  }
  __syncthreads();
  int far = 0;
  for (int it = 0; it < Nn; ++it) {
    if (tid == 0) fps_idx[b * Nn + it] = far;
    float cx = px[far], cy = py[far], cz = pz[far];
    float bd = -1.0f; int bi = 0;
    for (int n = tid; n < Nn; n += 256) {
      float dx = px[n] - cx, dy = py[n] - cy, dz = pz[n] - cz;
      float d = fminf(dist[n], dx * dx + dy * dy + dz * dz);
      dist[n] = d;
      if (d > bd || (d == bd && n < bi)) { bd = d; bi = n; }
    }
#pragma unroll
    for (int off = 16; off > 0; off >>= 1) {
      float od = __shfl_xor(bd, off, 32);
      int   oi = __shfl_xor(bi, off, 32);
      if (od > bd || (od == bd && oi < bi)) { bd = od; bi = oi; }
    }
    if (lane == 0) { wmx[wid] = bd; wmi[wid] = bi; }
    __syncthreads();
    if (tid == 0) {
      float bb = wmx[0]; int bj = wmi[0];
      for (int w = 1; w < 8; ++w)
        if (wmx[w] > bb || (wmx[w] == bb && wmi[w] < bj)) { bb = wmx[w]; bj = wmi[w]; }
      s_far = bj;
    }
    __syncthreads();
    far = s_far;
  }
}

__global__ void gather_newxyz_kernel(const float* __restrict__ xyz1t,
                                     const int* __restrict__ fps_idx,
                                     float* __restrict__ new_xyz) {
  int q = blockIdx.x * blockDim.x + threadIdx.x;
  if (q >= Bb * Nn) return;
  int b = q / Nn;
  int j = fps_idx[q];
  const float* p = &xyz1t[((size_t)b * Nn + j) * 3];
  new_xyz[q * 3 + 0] = p[0];
  new_xyz[q * 3 + 1] = p[1];
  new_xyz[q * 3 + 2] = p[2];
}

// ---------------- 16-NN selection: one wave per query, LDS slab ----------------
__global__ void knn_kernel(const float* __restrict__ xyz1t,
                           const float* __restrict__ new_xyz,
                           int* __restrict__ knn_idx) {
  __shared__ float dsl[8][Nn];
  int tid = threadIdx.x, lane = tid & 31, wid = tid >> 5;
  int q = blockIdx.x * 8 + wid;
  int b = q / Nn;
  float qx = new_xyz[q * 3 + 0], qy = new_xyz[q * 3 + 1], qz = new_xyz[q * 3 + 2];
  for (int m = lane; m < Nn; m += 32) {
    const float* p = &xyz1t[((size_t)b * Nn + m) * 3];
    float dx = qx - p[0], dy = qy - p[1], dz = qz - p[2];
    dsl[wid][m] = dx * dx + dy * dy + dz * dz;
  }
  __syncthreads();
  for (int r = 0; r < KNN; ++r) {
    float bd = 3.4e38f; int bi = 0;
    for (int m = lane; m < Nn; m += 32) {
      float d = dsl[wid][m];
      if (d < bd || (d == bd && m < bi)) { bd = d; bi = m; }
    }
#pragma unroll
    for (int off = 16; off > 0; off >>= 1) {
      float od = __shfl_xor(bd, off, 32);
      int   oi = __shfl_xor(bi, off, 32);
      if (od < bd || (od == bd && oi < bi)) { bd = od; bi = oi; }
    }
    if (lane == 0) {
      knn_idx[(size_t)q * KNN + r] = bi;
      dsl[wid][bi] = 3.4e38f;
    }
    __syncthreads();
  }
}

// ---------------- build mlp input act0 (CIN0P x COLS per batch) ----------------
__global__ void build_act0_kernel(const int* __restrict__ knn,
                                  const float* __restrict__ xyz1t,
                                  const float* __restrict__ new_xyz,
                                  const unsigned short* __restrict__ interp,
                                  const float* __restrict__ density_scale,
                                  unsigned short* __restrict__ act0,
                                  float* __restrict__ denscol) {
  int idx = blockIdx.x * blockDim.x + threadIdx.x;
  if (idx >= Bb * Nn * KNN) return;
  int b = idx / (Nn * KNN);
  int rem = idx - b * Nn * KNN;
  int n = rem / KNN, k = rem % KNN;
  int j = knn[((size_t)(b * Nn + n)) * KNN + k];
  int col = k * Nn + n;
  unsigned short* base = act0 + (size_t)b * CIN0P * COLS;
  const float* pj = &xyz1t[((size_t)b * Nn + j) * 3];
  const float* pq = &new_xyz[((size_t)b * Nn + n) * 3];
  base[(size_t)0 * COLS + col] = f2bf(pj[0] - pq[0]);
  base[(size_t)1 * COLS + col] = f2bf(pj[1] - pq[1]);
  base[(size_t)2 * COLS + col] = f2bf(pj[2] - pq[2]);
  const unsigned short* ip = interp + (size_t)(b * Nn + j) * Dd;
#pragma unroll 8
  for (int c = 0; c < Dd; ++c) base[(size_t)(3 + c) * COLS + col] = ip[c];
  for (int c = CIN0; c < CIN0P; ++c) base[(size_t)c * COLS + col] = 0;
  denscol[(size_t)b * COLS + col] = density_scale[b * Nn + j];
}

// ---------------- generic WMMA bf16 GEMM: Y = W @ X (+bias) ----------------
// LDS layouts (A natural K order, B transposed) are chosen so every WMMA operand fragment is two
// contiguous ds_load_b128 reads. Register-level double buffering overlaps the
// next K-slice's global loads with the WMMA stream; global_prefetch_b8 runs
// one slice further ahead.
template <bool OUT_F32>
__global__ void __launch_bounds__(256)
gemm_bf16_kernel(const unsigned short* __restrict__ Wp,  // (C_out x C_in) bf16
                 const unsigned short* __restrict__ X,   // (C_in x ncols) per batch
                 const float* __restrict__ bias,         // C_out or null
                 unsigned short* __restrict__ Yb,
                 float* __restrict__ Yf,
                 int C_out, int C_in, int ncols,
                 long long xBatchStride, long long yBatchStride) {
  // A: 64 rows x 32 K, natural K order per row, so the
  //    half-wave fragment (K=kb..kb+15, kb=16*half) is 16 contiguous shorts.
  // B: stored transposed: 128 cols x 32 K (natural K order); a lane fragment
  //    (K=kbase..kbase+15 at fixed col) is 16 contiguous shorts.
  __shared__ unsigned short Asw[64][40];   // 80B rows -> 16B-aligned fragments
  __shared__ unsigned short Bt[128][40];

  const int tid = threadIdx.x;
  const int lane = tid & 31;
  const int wid = tid >> 5;
  const int half = lane >> 4;
  const int rl = lane & 15;

  const int col0 = blockIdx.x * 128;
  const int co0 = blockIdx.y * 64;
  const int b = blockIdx.z;

  const unsigned short* Xb = X + (size_t)b * xBatchStride;
  const int m_base = (wid & 3) * 16;
  const int c_base = (wid >> 2) * 64;

  // cooperative-load geometry (uint4 = 8 bf16)
  const int ar = tid >> 2;                 // A row 0..63
  const int akg = (tid & 3) * 8;           // A k-group start (multiple of 8)
  const int apos = akg; // natural-K dst
  const int br0 = (tid * 8) >> 7;          // B rows for the two uint4 loads
  const int bc0 = (tid * 8) & 127;
  const int br1 = ((tid + 256) * 8) >> 7;
  const int bc1 = ((tid + 256) * 8) & 127;

  v8f acc[4];
#pragma unroll
  for (int s = 0; s < 4; ++s)
#pragma unroll
    for (int e = 0; e < 8; ++e) acc[s][e] = 0.0f;

  uint4 aReg, bReg0, bReg1;

  auto gload = [&](int kk) {
    aReg  = *(const uint4*)(Wp + (size_t)(co0 + ar) * C_in + kk + akg);
    bReg0 = *(const uint4*)(Xb + (size_t)(kk + br0) * ncols + col0 + bc0);
    bReg1 = *(const uint4*)(Xb + (size_t)(kk + br1) * ncols + col0 + bc1);
  };
  auto sstore = [&]() {
    *(uint4*)&Asw[ar][apos] = aReg;
    union { uint4 q; unsigned short s[8]; } u0, u1;
    u0.q = bReg0; u1.q = bReg1;
#pragma unroll
    for (int e = 0; e < 8; ++e) {
      Bt[bc0 + e][br0] = u0.s[e];
      Bt[bc1 + e][br1] = u1.s[e];
    }
  };
  auto compute = [&]() {
    union { v16bf v; uint4 q[2]; } af;
    {
      const uint4* ap = (const uint4*)&Asw[m_base + rl][half * 16];
      af.q[0] = ap[0];
      af.q[1] = ap[1];
    }
#pragma unroll
    for (int s = 0; s < 4; ++s) {
      union { v16bf v; uint4 q[2]; } bfr;
      const uint4* bp = (const uint4*)&Bt[c_base + s * 16 + rl][half * 16];
      bfr.q[0] = bp[0];
      bfr.q[1] = bp[1];
      acc[s] = __builtin_amdgcn_wmma_f32_16x16x32_bf16(
          false, af.v, false, bfr.v, (short)0, acc[s], false, false);
    }
  };

  gload(0);
  sstore();
  __syncthreads();
  for (int kk = 32; kk < C_in; kk += 32) {
    gload(kk);                              // next slice -> registers
    if (kk + 32 < C_in) {                   // gfx1250 global_prefetch_b8
      __builtin_prefetch(Xb + (size_t)(kk + 32 + br0) * ncols + col0 + bc0, 0, 0);
    }
    compute();                              // WMMA on current LDS tiles
    __syncthreads();
    sstore();                               // registers -> LDS
    __syncthreads();
  }
  compute();

#pragma unroll
  for (int s = 0; s < 4; ++s) {
#pragma unroll
    for (int v = 0; v < 8; ++v) {
      int grow = co0 + m_base + half * 8 + v;
      int gcol = col0 + c_base + s * 16 + rl;
      float val = acc[s][v] + (bias ? bias[grow] : 0.0f);
      if constexpr (OUT_F32)
        Yf[(size_t)b * yBatchStride + (size_t)grow * ncols + gcol] = val;
      else
        Yb[(size_t)b * yBatchStride + (size_t)grow * ncols + gcol] = f2bf(val);
    }
  }
}

// ---------------- per-(batch,channel) mean/var ----------------
template <bool BF16IN>
__global__ void colstats_kernel(const void* __restrict__ Xv, float* __restrict__ stats,
                                int C, long long ncols, long long batchStride) {
  __shared__ float r1[256], r2[256];
  int bid = blockIdx.x;
  int bq = bid / C, cq = bid % C;
  const unsigned short* Xb = (const unsigned short*)Xv;
  const float* Xf = (const float*)Xv;
  size_t base = (size_t)bq * batchStride + (size_t)cq * ncols;
  float s1 = 0.f, s2 = 0.f;
  for (long long i = threadIdx.x; i < ncols; i += 256) {
    float x = BF16IN ? bf2f(Xb[base + i]) : Xf[base + i];
    s1 += x; s2 += x * x;
  }
  r1[threadIdx.x] = s1; r2[threadIdx.x] = s2;
  __syncthreads();
  for (int o = 128; o > 0; o >>= 1) {
    if (threadIdx.x < o) { r1[threadIdx.x] += r1[threadIdx.x + o]; r2[threadIdx.x] += r2[threadIdx.x + o]; }
    __syncthreads();
  }
  if (threadIdx.x == 0) {
    float m = r1[0] / (float)ncols;
    float v = r2[0] / (float)ncols - m * m;
    stats[bid * 2 + 0] = m;
    stats[bid * 2 + 1] = v;
  }
}

// ---------------- normalize + relu (+optional density) in-place, bf16 ----------------
__global__ void normrelu_kernel(unsigned short* __restrict__ X,
                                const float* __restrict__ stats,
                                const float* __restrict__ dens,
                                int C, long long ncols, long long batchStride,
                                long long densStride, long long total) {
  long long idx = (long long)blockIdx.x * blockDim.x + threadIdx.x;
  long long stride = (long long)gridDim.x * blockDim.x;
  for (; idx < total; idx += stride) {
    long long b = idx / ((long long)C * ncols);
    long long rem = idx - b * (long long)C * ncols;
    int c = (int)(rem / ncols);
    long long col = rem - (long long)c * ncols;
    size_t pos = (size_t)b * batchStride + (size_t)c * ncols + col;
    float m = stats[(b * C + c) * 2 + 0];
    float v = stats[(b * C + c) * 2 + 1];
    float y = (bf2f(X[pos]) - m) * rsqrtf(v + EPS);
    y = fmaxf(y, 0.0f);
    if (dens) y *= dens[(size_t)b * densStride + col];
    X[pos] = f2bf(y);
  }
}

// ---------------- small conv1x1 for the weight net (CI<=8, CO<=16) ----------------
__global__ void smallconv_kernel(const float* __restrict__ W, const float* __restrict__ bias,
                                 const unsigned short* __restrict__ in,
                                 unsigned short* __restrict__ out,
                                 int CO, int CI, long long ncols,
                                 long long inBatchStride, long long outBatchStride) {
  long long idx = (long long)blockIdx.x * blockDim.x + threadIdx.x;
  long long total = (long long)Bb * ncols;
  for (; idx < total; idx += (long long)gridDim.x * blockDim.x) {
    long long b = idx / ncols;
    long long col = idx - b * ncols;
    const unsigned short* ib = in + (size_t)b * inBatchStride;
    unsigned short* ob = out + (size_t)b * outBatchStride;
    float xi[8];
    for (int ci = 0; ci < CI; ++ci) xi[ci] = bf2f(ib[(size_t)ci * ncols + col]);
    for (int co = 0; co < CO; ++co) {
      float s = bias[co];
      for (int ci = 0; ci < CI; ++ci) s += W[co * CI + ci] * xi[ci];
      ob[(size_t)co * ncols + col] = f2bf(s);
    }
  }
}

// ---------------- einsum bckn,bwkn->bncw -> X_lin[i, b*N+n] ----------------
__global__ void einsum_kernel(const unsigned short* __restrict__ act3,
                              const unsigned short* __restrict__ wts,
                              unsigned short* __restrict__ outpre) {
  __shared__ unsigned short featS[256][KNN];
  __shared__ unsigned short wtsS[16][KNN];
  int p = blockIdx.x;                       // b*N+n
  int b = p / Nn, n = p % Nn;
  int tid = threadIdx.x;
  { // load feat (256 x 16)
    int c = tid;
#pragma unroll
    for (int k = 0; k < KNN; ++k)
      featS[c][k] = act3[(size_t)(b * COUT3 + c) * COLS + (size_t)k * Nn + n];
  }
  if (tid < 16) {
    int w = tid;
#pragma unroll
    for (int k = 0; k < KNN; ++k)
      wtsS[w][k] = wts[(size_t)(b * 16 + w) * COLS + (size_t)k * Nn + n];
  }
  __syncthreads();
  int c = tid;
  float f[KNN];
#pragma unroll
  for (int k = 0; k < KNN; ++k) f[k] = bf2f(featS[c][k]);
#pragma unroll
  for (int w = 0; w < 16; ++w) {
    float s = 0.f;
#pragma unroll
    for (int k = 0; k < KNN; ++k) s += f[k] * bf2f(wtsS[w][k]);
    outpre[(size_t)(c * 16 + w) * FCOLS + p] = f2bf(s);
  }
}

// ---------------- final inorm+relu -> d_out (B,N,256) ----------------
__global__ void finnorm_kernel(const float* __restrict__ yfinal,
                               const float* __restrict__ stats,
                               float* __restrict__ dout) {
  long long idx = (long long)blockIdx.x * blockDim.x + threadIdx.x;
  if (idx >= (long long)Bb * Nn * 256) return;
  int o = (int)(idx & 255);
  long long col2 = idx >> 8;                // b*N+n
  int b = (int)(col2 >> 11);                // N = 2048
  float m = stats[(o * Bb + b) * 2 + 0];
  float v = stats[(o * Bb + b) * 2 + 1];
  float x = yfinal[(size_t)o * FCOLS + col2];
  float y = fmaxf((x - m) * rsqrtf(v + EPS), 0.0f);
  dout[(size_t)col2 * 256 + o] = y;
}

// ==================== host orchestration ====================
extern "C" void kernel_launch(void* const* d_in, const int* in_sizes, int n_in,
                              void* d_out, int out_size, void* d_ws, size_t ws_size,
                              hipStream_t stream) {
  (void)in_sizes; (void)n_in; (void)out_size; (void)ws_size;
  const float* xyz1 = (const float*)d_in[0];
  const float* xyz2 = (const float*)d_in[1];
  const float* points2 = (const float*)d_in[3];
  const float* dn_w0 = (const float*)d_in[4];  const float* dn_b0 = (const float*)d_in[5];
  const float* dn_w1 = (const float*)d_in[6];  const float* dn_b1 = (const float*)d_in[7];
  const float* dn_w2 = (const float*)d_in[8];  const float* dn_b2 = (const float*)d_in[9];
  const float* wn_w0 = (const float*)d_in[10]; const float* wn_b0 = (const float*)d_in[11];
  const float* wn_w1 = (const float*)d_in[12]; const float* wn_b1 = (const float*)d_in[13];
  const float* wn_w2 = (const float*)d_in[14]; const float* wn_b2 = (const float*)d_in[15];
  const float* mlp_w0 = (const float*)d_in[16]; const float* mlp_b0 = (const float*)d_in[17];
  const float* mlp_w1 = (const float*)d_in[18]; const float* mlp_b1 = (const float*)d_in[19];
  const float* mlp_w2 = (const float*)d_in[20]; const float* mlp_b2 = (const float*)d_in[21];
  const float* lin_w = (const float*)d_in[22];  const float* lin_b = (const float*)d_in[23];
  float* dout = (float*)d_out;

  char* base = (char*)d_ws; size_t off = 0;
  auto alloc = [&](size_t bytes) -> void* {
    off = (off + 255) & ~(size_t)255;
    void* p = base + off; off += bytes; return p;
  };

  float* xyz1t = (float*)alloc((size_t)Bb * Nn * 3 * 4);
  float* xyz2t = (float*)alloc((size_t)Bb * Ss * 3 * 4);
  unsigned short* interp = (unsigned short*)alloc((size_t)Bb * Nn * Dd * 2);
  float* density = (float*)alloc((size_t)Bb * Nn * 4);
  float* density_scale = (float*)alloc((size_t)Bb * Nn * 4);
  int* fps_idx = (int*)alloc((size_t)Bb * Nn * 4);
  float* new_xyz = (float*)alloc((size_t)Bb * Nn * 3 * 4);
  int* knn = (int*)alloc((size_t)Bb * Nn * KNN * 4);
  float* denscol = (float*)alloc((size_t)Bb * COLS * 4);
  unsigned short* w0p = (unsigned short*)alloc((size_t)CMID * CIN0P * 2);
  unsigned short* w1p = (unsigned short*)alloc((size_t)CMID * CMID * 2);
  unsigned short* w2p = (unsigned short*)alloc((size_t)COUT3 * CMID * 2);
  unsigned short* wlp = (unsigned short*)alloc((size_t)LIN_OUT * LIN_IN * 2);
  float* stats = (float*)alloc((size_t)4096 * 2 * 4);
  unsigned short* wtsA = (unsigned short*)alloc((size_t)Bb * 8 * COLS * 2);
  unsigned short* wtsB = (unsigned short*)alloc((size_t)Bb * 8 * COLS * 2);
  unsigned short* wtsF = (unsigned short*)alloc((size_t)Bb * 16 * COLS * 2);
  unsigned short* act0 = (unsigned short*)alloc((size_t)Bb * CIN0P * COLS * 2);  // reused as act3
  unsigned short* act12 = (unsigned short*)alloc((size_t)2 * Bb * CMID * COLS * 2); // act1|act2, reused as outpre
  float* yfinal = (float*)alloc((size_t)LIN_OUT * FCOLS * 4);

  unsigned short* act1 = act12;
  unsigned short* act2 = act12 + (size_t)Bb * CMID * COLS;
  unsigned short* act3 = act0;
  unsigned short* outpre = act12;

  // 1. weights -> bf16
  prep_weights_kernel<<<4096, 256, 0, stream>>>(mlp_w0, mlp_w1, mlp_w2, lin_w, w0p, w1p, w2p, wlp);
  // 2. transpose
  transpose_kernel<<<240, 256, 0, stream>>>(xyz1, xyz2, xyz1t, xyz2t);
  // 3. 3-NN interpolation of points2
  interp_kernel<<<Bb * Nn / 8, 256, 0, stream>>>(xyz1t, xyz2t, points2, interp);
  // 4. kernel density
  density_kernel<<<Bb * Nn / 8, 256, 0, stream>>>(xyz1t, density);
  // 5. density net
  dn_net_kernel<<<Bb, 256, 0, stream>>>(density, dn_w0, dn_b0, dn_w1, dn_b1, dn_w2, dn_b2, density_scale);
  // 6. farthest point sampling
  fps_kernel<<<Bb, 256, 0, stream>>>(xyz1t, fps_idx);
  gather_newxyz_kernel<<<(Bb * Nn + 255) / 256, 256, 0, stream>>>(xyz1t, fps_idx, new_xyz);
  // 7. 16-NN
  knn_kernel<<<Bb * Nn / 8, 256, 0, stream>>>(xyz1t, new_xyz, knn);
  // 8. build act0 + gathered density
  build_act0_kernel<<<(Bb * Nn * KNN + 255) / 256, 256, 0, stream>>>(
      knn, xyz1t, new_xyz, interp, density_scale, act0, denscol);

  // 9. weight net (3->8->8->16) on channels 0..2 of act0
  smallconv_kernel<<<1024, 256, 0, stream>>>(wn_w0, wn_b0, act0, wtsA, 8, 3, COLS,
                                             (long long)CIN0P * COLS, (long long)8 * COLS);
  colstats_kernel<true><<<Bb * 8, 256, 0, stream>>>(wtsA, stats, 8, COLS, (long long)8 * COLS);
  normrelu_kernel<<<8192, 256, 0, stream>>>(wtsA, stats, nullptr, 8, COLS, (long long)8 * COLS, 0,
                                            (long long)Bb * 8 * COLS);
  smallconv_kernel<<<1024, 256, 0, stream>>>(wn_w1, wn_b1, wtsA, wtsB, 8, 8, COLS,
                                             (long long)8 * COLS, (long long)8 * COLS);
  colstats_kernel<true><<<Bb * 8, 256, 0, stream>>>(wtsB, stats, 8, COLS, (long long)8 * COLS);
  normrelu_kernel<<<8192, 256, 0, stream>>>(wtsB, stats, nullptr, 8, COLS, (long long)8 * COLS, 0,
                                            (long long)Bb * 8 * COLS);
  smallconv_kernel<<<1024, 256, 0, stream>>>(wn_w2, wn_b2, wtsB, wtsF, 16, 8, COLS,
                                             (long long)8 * COLS, (long long)16 * COLS);
  colstats_kernel<true><<<Bb * 16, 256, 0, stream>>>(wtsF, stats, 16, COLS, (long long)16 * COLS);
  normrelu_kernel<<<16384, 256, 0, stream>>>(wtsF, stats, nullptr, 16, COLS, (long long)16 * COLS, 0,
                                             (long long)Bb * 16 * COLS);

  // 10. mlp layer 0: 288 -> 128   (WMMA)
  {
    dim3 g(COLS / 128, CMID / 64, Bb);
    gemm_bf16_kernel<false><<<g, 256, 0, stream>>>(w0p, act0, mlp_b0, act1, nullptr,
                                                   CMID, CIN0P, COLS,
                                                   (long long)CIN0P * COLS, (long long)CMID * COLS);
  }
  colstats_kernel<true><<<Bb * CMID, 256, 0, stream>>>(act1, stats, CMID, COLS, (long long)CMID * COLS);
  normrelu_kernel<<<65536, 256, 0, stream>>>(act1, stats, nullptr, CMID, COLS, (long long)CMID * COLS, 0,
                                             (long long)Bb * CMID * COLS);
  // 11. mlp layer 1: 128 -> 128
  {
    dim3 g(COLS / 128, CMID / 64, Bb);
    gemm_bf16_kernel<false><<<g, 256, 0, stream>>>(w1p, act1, mlp_b1, act2, nullptr,
                                                   CMID, CMID, COLS,
                                                   (long long)CMID * COLS, (long long)CMID * COLS);
  }
  colstats_kernel<true><<<Bb * CMID, 256, 0, stream>>>(act2, stats, CMID, COLS, (long long)CMID * COLS);
  normrelu_kernel<<<65536, 256, 0, stream>>>(act2, stats, nullptr, CMID, COLS, (long long)CMID * COLS, 0,
                                             (long long)Bb * CMID * COLS);
  // 12. mlp layer 2: 128 -> 256, then norm+relu and fold in grouped density
  {
    dim3 g(COLS / 128, COUT3 / 64, Bb);
    gemm_bf16_kernel<false><<<g, 256, 0, stream>>>(w2p, act2, mlp_b2, act3, nullptr,
                                                   COUT3, CMID, COLS,
                                                   (long long)CMID * COLS, (long long)COUT3 * COLS);
  }
  colstats_kernel<true><<<Bb * COUT3, 256, 0, stream>>>(act3, stats, COUT3, COLS, (long long)COUT3 * COLS);
  normrelu_kernel<<<131072, 256, 0, stream>>>(act3, stats, denscol, COUT3, COLS,
                                              (long long)COUT3 * COLS, (long long)COLS,
                                              (long long)Bb * COUT3 * COLS);

  // 13. einsum -> outpre (4096 x 16384)
  einsum_kernel<<<Bb * Nn, 256, 0, stream>>>(act3, wtsF, outpre);

  // 14. final linear 4096 -> 256 (WMMA, f32 out)
  {
    dim3 g(FCOLS / 128, LIN_OUT / 64, 1);
    gemm_bf16_kernel<true><<<g, 256, 0, stream>>>(wlp, outpre, lin_b, nullptr, yfinal,
                                                  LIN_OUT, LIN_IN, FCOLS, 0, 0);
  }
  // 15. final inorm over N per (b, channel): stats base = o*FCOLS + b*N
  colstats_kernel<false><<<LIN_OUT * Bb, 256, 0, stream>>>(yfinal, stats, Bb, Nn, (long long)FCOLS);
  finnorm_kernel<<<(Bb * Nn * 256 + 255) / 256, 256, 0, stream>>>(yfinal, stats, dout);
}
